// TARelationConv_42142219108838
// MI455X (gfx1250) — matrix operand order repaired and checked by reading, not compiled, not run on hardware
//
#include <hip/hip_runtime.h>
#include <math.h>

// ---------------------------------------------------------------------------
// MI455X (gfx1250) implementation of TARelationConv.
// ~16 GFLOP compute vs ~25MB data -> compute bound: use v_wmma_f32_16x16x32_f16
// for every GEMM-shaped piece (MLPs, attention logits, attention apply, rel L2),
// f32 accumulate. KNN / K=10 rel layer1 / softmax stay on VALU (negligible).
// GEMM waves compute 16x128 strips: A fragment reused across 8 WMMAs.
// ---------------------------------------------------------------------------

typedef __attribute__((ext_vector_type(16))) _Float16 v16h;
typedef __attribute__((ext_vector_type(8)))  float    v8f;
typedef _Float16 h16;

#define WMMA_F16(a, b, c) \
  __builtin_amdgcn_wmma_f32_16x16x32_f16(false, (a), false, (b), (short)0, (c), false, false)

constexpr int S_  = 64;
constexpr int NO  = 256;   // objects per sentence
constexpr int LL  = 32;    // language tokens
constexpr int DD  = 128;   // feature dim
constexpr int LID = 256;   // language input dim
constexpr int K1  = 17;    // k + 1 neighbors (incl. self)

// ---------------------------------------------------------------------------
// 17-NN per object. (d, j) lexicographic selection matches top_k(-d) ordering.
// ---------------------------------------------------------------------------
__global__ void knn_kernel(const float* __restrict__ coord, int* __restrict__ idx)
{
    int t = blockIdx.x * blockDim.x + threadIdx.x;   // 0 .. S*NO-1
    int s = t >> 8, n = t & (NO - 1);
    const float* cs = coord + (size_t)s * NO * 3;
    float cx = cs[n * 3 + 0], cy = cs[n * 3 + 1], cz = cs[n * 3 + 2];
    float last_d = -1.0f; int last_j = -1;
    for (int r = 0; r < K1; ++r) {
        float best = 3.0e38f; int bj = 0;
        for (int j = 0; j < NO; ++j) {
            float dx = cs[j * 3 + 0] - cx;
            float dy = cs[j * 3 + 1] - cy;
            float dz = cs[j * 3 + 2] - cz;
            float d = dx * dx + dy * dy + dz * dz;
            bool after = (d > last_d) || (d == last_d && j > last_j);
            if (after && d < best) { best = d; bj = j; }
        }
        idx[(size_t)t * K1 + r] = bj;
        last_d = best; last_j = bj;
    }
}

// ---------------------------------------------------------------------------
// Weight convert f32 -> f16, transposed to [N][K] so WMMA B loads are contiguous.
// ---------------------------------------------------------------------------
__global__ void transpose_w_kernel(const float* __restrict__ src, h16* __restrict__ dst,
                                   int K, int N)
{
    int t = blockIdx.x * blockDim.x + threadIdx.x;
    if (t < K * N) {
        int k = t / N, n = t - k * N;
        dst[(size_t)n * K + k] = (h16)src[t];
    }
}

// langT[s][d][l] = lang[s][l][d]  (B fragment for attn-apply GEMM)
__global__ void lang_transpose_kernel(const h16* __restrict__ langf, h16* __restrict__ langT)
{
    int t = blockIdx.x * blockDim.x + threadIdx.x;   // S*LL*DD
    if (t < S_ * LL * DD) {
        int s = t >> 12; int rem = t & 4095; int l = rem >> 7; int d = rem & 127;
        langT[((size_t)s * DD + d) * LL + l] = langf[t];
    }
}

// ---------------------------------------------------------------------------
// One wave computes a 16x128 strip of  C = act(A[M,K] * Wt^T + bias), Nd = 128.
// A row-major f32 or f16; Wt is [128][K] f16 (transposed weights).
// A-fragment loaded once per K-step, reused by 8 WMMAs (8 accumulators).
// ---------------------------------------------------------------------------
template <bool RELU, typename AT>
__global__ void __launch_bounds__(32)
gemm16x128_kernel(const AT* __restrict__ A, const h16* __restrict__ Wt,
                  const float* __restrict__ bias, int M, int Kd,
                  h16* __restrict__ outH, float* __restrict__ outF)
{
    int lane = threadIdx.x & 31;
    int lo = lane & 15, hi = lane >> 4;
    int mt = blockIdx.x;
    int m = mt * 16 + lo;
    const AT* ap = A + (size_t)m * Kd;

    v8f acc[8];
#pragma unroll
    for (int t = 0; t < 8; ++t) {
        float bv = bias[t * 16 + lo];
#pragma unroll
        for (int v = 0; v < 8; ++v) acc[t][v] = bv;
    }

    for (int k0 = 0; k0 < Kd; k0 += 32) {
        if (k0 + 32 < Kd)
            __builtin_prefetch(ap + k0 + 32, 0, 1);   // global_prefetch_b8
        v16h a;
#pragma unroll
        for (int i = 0; i < 8; ++i) {
            a[i]     = (h16)ap[k0 + hi * 16 + i];          // K = k0+hi*16 .. +7
            a[i + 8] = (h16)ap[k0 + hi * 16 + 8 + i];     // K = k0+hi*16+8 .. +7
        }
#pragma unroll
        for (int t = 0; t < 8; ++t) {
            const h16* wp = Wt + (size_t)(t * 16 + lo) * Kd + k0 + hi * 16;
            v16h b;
#pragma unroll
            for (int i = 0; i < 16; ++i) b[i] = wp[i];
            acc[t] = WMMA_F16(a, b, acc[t]);
        }
    }

#pragma unroll
    for (int t = 0; t < 8; ++t) {
#pragma unroll
        for (int v = 0; v < 8; ++v) {
            int row = mt * 16 + hi * 8 + v;
            float val = acc[t][v];
            if (RELU) val = fmaxf(val, 0.0f);
            size_t off = (size_t)row * DD + t * 16 + lo;
            if (outH) outH[off] = (h16)val;
            if (outF) outF[off] = val;
        }
    }
}

// ---------------------------------------------------------------------------
// Per (s,n): rel layer1 (VALU) -> LDS, WMMA logits, softmax+mask+renorm,
// WMMA ins & rel layer2 per 16-col tile, fused sum_k f_nb*ins*rel + f, score.
// 64 threads = 2 waves; wave w owns neighbor rows [16w, 16w+16).
// ---------------------------------------------------------------------------
__global__ void __launch_bounds__(64)
combine_kernel(const float* __restrict__ coord, const int* __restrict__ idx,
               const int* __restrict__ lang_len,
               const float* __restrict__ rel_w1, const float* __restrict__ rel_b1,
               const h16* __restrict__ rel_w2t, const float* __restrict__ rel_b2,
               const float* __restrict__ f_f32, const h16* __restrict__ f_f16,
               const h16* __restrict__ lang_f16, const h16* __restrict__ langT,
               float* __restrict__ out, float* __restrict__ score)
{
    __shared__ h16   sm_hrel[32][DD];    // relu(rel10 @ W1 + b1), rows>=K1 zero
    __shared__ h16   sm_fnb[32][DD];     // gathered f16 features, rows>=K1 zero
    __shared__ float sm_logits[32][LL];
    __shared__ h16   sm_attn[32][LL];
    __shared__ float sm_part[2][DD];
    __shared__ float sm_score[64];

    int sn = blockIdx.x;
    int s = sn >> 8, n = sn & (NO - 1);
    int tid = threadIdx.x;
    int lane = tid & 31, wave = tid >> 5;
    int lo = lane & 15, hi = lane >> 4;
    int mrow = wave * 16 + lo;

    const float* cs = coord + (size_t)s * NO * 3;
    float cx = cs[n * 3 + 0], cy = cs[n * 3 + 1], cz = cs[n * 3 + 2];
    const int* idxp = idx + (size_t)sn * K1;

    // ---- rel layer1 (K=10, VALU) + neighbor feature gather into LDS --------
    for (int kk = 0; kk < K1; ++kk) {
        int j = idxp[kk];
        float nx = cs[j * 3 + 0], ny = cs[j * 3 + 1], nz = cs[j * 3 + 2];
        float rx = nx - cx, ry = ny - cy, rz = nz - cz;
        float dist = sqrtf(rx * rx + ry * ry + rz * rz);
        float fe[10] = {nx, ny, nz, cx, cy, cz, rx, ry, rz, dist};
        const h16* frow = f_f16 + ((size_t)s * NO + j) * DD;
        for (int d = tid; d < DD; d += 64) {
            float h = rel_b1[d];
#pragma unroll
            for (int i = 0; i < 10; ++i) h += fe[i] * rel_w1[i * DD + d];
            sm_hrel[kk][d] = (h16)fmaxf(h, 0.0f);
            sm_fnb[kk][d]  = frow[d];
        }
    }
    for (int kk = K1; kk < 32; ++kk)
        for (int d = tid; d < DD; d += 64) { sm_hrel[kk][d] = (h16)0.0f; sm_fnb[kk][d] = (h16)0.0f; }
    __syncthreads();

    // ---- logits = f_nb (17x128) @ lang^T (128x32) -> WMMA ------------------
    {
        v8f accL0 = {}, accL1 = {};
        const h16* ap = &sm_fnb[mrow][0];
#pragma unroll
        for (int ks = 0; ks < 4; ++ks) {
            int k0 = ks * 32;
            v16h a, b0, b1;
#pragma unroll
            for (int i = 0; i < 8; ++i) {
                a[i]     = ap[k0 + hi * 16 + i];
                a[i + 8] = ap[k0 + hi * 16 + 8 + i];
            }
            const h16* lp0 = lang_f16 + ((size_t)s * LL + lo)      * DD + k0 + hi * 16;
            const h16* lp1 = lang_f16 + ((size_t)s * LL + 16 + lo) * DD + k0 + hi * 16;
#pragma unroll
            for (int i = 0; i < 16; ++i) { b0[i] = lp0[i]; b1[i] = lp1[i]; }
            accL0 = WMMA_F16(a, b0, accL0);
            accL1 = WMMA_F16(a, b1, accL1);
        }
#pragma unroll
        for (int v = 0; v < 8; ++v) {
            int r = wave * 16 + hi * 8 + v;
            sm_logits[r][lo]      = accL0[v];
            sm_logits[r][16 + lo] = accL1[v];
        }
    }
    __syncthreads();

    // ---- softmax over l, mask by lang_len, renormalize (+1e-7) -------------
    if (tid < K1) {
        float e[LL];
        float mx = -3.0e38f;
#pragma unroll
        for (int l = 0; l < LL; ++l) mx = fmaxf(mx, sm_logits[tid][l]);
        float sum = 0.0f;
#pragma unroll
        for (int l = 0; l < LL; ++l) { e[l] = __expf(sm_logits[tid][l] - mx); sum += e[l]; }
        int len = lang_len[s];
        float msum = 0.0f;
#pragma unroll
        for (int l = 0; l < LL; ++l) {
            float p = e[l] / sum;
            p = (l < len) ? p : 0.0f;
            e[l] = p; msum += p;
        }
        float inv = 1.0f / (msum + 1e-7f);
#pragma unroll
        for (int l = 0; l < LL; ++l) sm_attn[tid][l] = (h16)(e[l] * inv);
    } else if (tid < 32) {
#pragma unroll
        for (int l = 0; l < LL; ++l) sm_attn[tid][l] = (h16)0.0f;
    }
    __syncthreads();

    // ---- per 16-col tile: rel layer2 WMMA + ins WMMA + fused combine -------
    for (int t = 0; t < 8; ++t) {
        int d = t * 16 + lo;

        // rel2: h_rel (17x128) @ W2 (128x128) tile, bias splat init
        float bv = rel_b2[d];
        v8f accR;
#pragma unroll
        for (int v = 0; v < 8; ++v) accR[v] = bv;
        const h16* hp = &sm_hrel[mrow][0];
        const h16* wp = rel_w2t + (size_t)d * DD;
#pragma unroll
        for (int ks = 0; ks < 4; ++ks) {
            int k0 = ks * 32;
            v16h a, b;
#pragma unroll
            for (int i = 0; i < 8; ++i) {
                a[i]     = hp[k0 + hi * 16 + i];
                a[i + 8] = hp[k0 + hi * 16 + 8 + i];
            }
#pragma unroll
            for (int i = 0; i < 16; ++i) b[i] = wp[k0 + hi * 16 + i];
            accR = WMMA_F16(a, b, accR);
        }

        // ins: attn (17x32) @ lang (32x128) tile, single K-step
        v8f accI = {};
        {
            const h16* atp = &sm_attn[mrow][0];
            v16h a, b;
#pragma unroll
            for (int i = 0; i < 8; ++i) {
                a[i]     = atp[hi * 16 + i];
                a[i + 8] = atp[hi * 16 + 8 + i];
            }
            const h16* ltp = langT + ((size_t)s * DD + d) * LL + hi * 16;
#pragma unroll
            for (int i = 0; i < 16; ++i) b[i] = ltp[i];
            accI = WMMA_F16(a, b, accI);
        }

        // sum over this wave's neighbor rows: f_nb * ins * rel
        float p = 0.0f;
#pragma unroll
        for (int v = 0; v < 8; ++v) {
            int kk = wave * 16 + hi * 8 + v;
            if (kk < K1) p += (float)sm_fnb[kk][d] * accI[v] * accR[v];
        }
        p += __shfl_xor(p, 16, 32);          // lane L & L+16 share column d
        if (hi == 0) sm_part[wave][d] = p;
    }
    __syncthreads();

    // ---- out = sum_k + f, score = row sum ----------------------------------
    float my = 0.0f;
    for (int d = tid; d < DD; d += 64) {
        float o = sm_part[0][d] + sm_part[1][d] + f_f32[(size_t)sn * DD + d];
        out[(size_t)sn * DD + d] = o;
        my += o;
    }
    sm_score[tid] = my;
    __syncthreads();
    if (tid == 0) {
        float sc = 0.0f;
        for (int i = 0; i < 64; ++i) sc += sm_score[i];
        score[sn] = sc;
    }
}

// ---------------------------------------------------------------------------
extern "C" void kernel_launch(void* const* d_in, const int* in_sizes, int n_in,
                              void* d_out, int out_size, void* d_ws, size_t ws_size,
                              hipStream_t stream)
{
    (void)in_sizes; (void)n_in; (void)out_size; (void)ws_size;

    const float* feat      = (const float*)d_in[0];
    const float* coord     = (const float*)d_in[1];
    const float* lang_feat = (const float*)d_in[2];
    const int*   lang_len  = (const int*)  d_in[3];
    const float* rel_w1    = (const float*)d_in[4];
    const float* rel_b1    = (const float*)d_in[5];
    const float* rel_w2    = (const float*)d_in[6];
    const float* rel_b2    = (const float*)d_in[7];
    const float* lang_w1   = (const float*)d_in[8];
    const float* lang_b1   = (const float*)d_in[9];
    const float* lang_w2   = (const float*)d_in[10];
    const float* lang_b2   = (const float*)d_in[11];
    const float* feat_w1   = (const float*)d_in[12];
    const float* feat_b1   = (const float*)d_in[13];
    const float* feat_w2   = (const float*)d_in[14];
    const float* feat_b2   = (const float*)d_in[15];

    float* out   = (float*)d_out;                 // [S,N,D]
    float* score = out + (size_t)S_ * NO * DD;    // [S,N]

    // workspace carve-out (~19 MB)
    char* wptr = (char*)d_ws;
    auto alloc = [&](size_t bytes) -> void* {
        void* p = (void*)wptr;
        wptr += (bytes + 255) & ~(size_t)255;
        return p;
    };
    int* idx       = (int*)alloc((size_t)S_ * NO * K1 * sizeof(int));
    h16* lang_w1t  = (h16*)alloc((size_t)DD * LID * 2);
    h16* lang_w2t  = (h16*)alloc((size_t)DD * DD * 2);
    h16* feat_w1t  = (h16*)alloc((size_t)DD * DD * 2);
    h16* feat_w2t  = (h16*)alloc((size_t)DD * DD * 2);
    h16* rel_w2t   = (h16*)alloc((size_t)DD * DD * 2);
    h16* lang_h1   = (h16*)alloc((size_t)S_ * LL * DD * 2);
    h16* lang_f16  = (h16*)alloc((size_t)S_ * LL * DD * 2);
    h16* langT     = (h16*)alloc((size_t)S_ * DD * LL * 2);
    h16* feat_h1   = (h16*)alloc((size_t)S_ * NO * DD * 2);
    h16* f_f16     = (h16*)alloc((size_t)S_ * NO * DD * 2);
    float* f_f32   = (float*)alloc((size_t)S_ * NO * DD * 4);

    // 1) KNN
    knn_kernel<<<S_, NO, 0, stream>>>(coord, idx);

    // 2) weights -> f16 transposed
    transpose_w_kernel<<<(LID * DD + 255) / 256, 256, 0, stream>>>(lang_w1, lang_w1t, LID, DD);
    transpose_w_kernel<<<(DD * DD + 255) / 256, 256, 0, stream>>>(lang_w2, lang_w2t, DD, DD);
    transpose_w_kernel<<<(DD * DD + 255) / 256, 256, 0, stream>>>(feat_w1, feat_w1t, DD, DD);
    transpose_w_kernel<<<(DD * DD + 255) / 256, 256, 0, stream>>>(feat_w2, feat_w2t, DD, DD);
    transpose_w_kernel<<<(DD * DD + 255) / 256, 256, 0, stream>>>(rel_w2, rel_w2t, DD, DD);

    // 3) lang MLP: [2048,256] -> relu[2048,128] -> [2048,128] (f16) + transpose
    gemm16x128_kernel<true, float><<<(S_ * LL) / 16, 32, 0, stream>>>(
        lang_feat, lang_w1t, lang_b1, S_ * LL, LID, lang_h1, nullptr);
    gemm16x128_kernel<false, h16><<<(S_ * LL) / 16, 32, 0, stream>>>(
        lang_h1, lang_w2t, lang_b2, S_ * LL, DD, lang_f16, nullptr);
    lang_transpose_kernel<<<(S_ * LL * DD + 255) / 256, 256, 0, stream>>>(lang_f16, langT);

    // 4) feat MLP: [16384,128] -> relu -> f (f32 + f16)
    gemm16x128_kernel<true, float><<<(S_ * NO) / 16, 32, 0, stream>>>(
        feat, feat_w1t, feat_b1, S_ * NO, DD, feat_h1, nullptr);
    gemm16x128_kernel<false, h16><<<(S_ * NO) / 16, 32, 0, stream>>>(
        feat_h1, feat_w2t, feat_b2, S_ * NO, DD, f_f16, f_f32);

    // 5) fused rel-MLP + attention + combine, one block per (s,n)
    combine_kernel<<<S_ * NO, 64, 0, stream>>>(
        coord, idx, lang_len, rel_w1, rel_b1, rel_w2t, rel_b2,
        f_f32, f_f16, lang_f16, langT, out, score);
}
